// NanoRAG_80951543595477
// MI455X (gfx1250) — matrix operand order repaired and checked by reading, not compiled, not run on hardware
//
#include <hip/hip_runtime.h>
#include <hip/hip_bf16.h>
#include <math.h>

typedef __attribute__((ext_vector_type(16))) __bf16 v16bf;
typedef __attribute__((ext_vector_type(8)))  float  v8f;

#define EMBED   384
#define NDOCS   262144
#define BATCH   64
#define KSEL    8
#define SBLOCKS 256
#define DPB     (NDOCS / SBLOCKS)   /* 1024 docs per scoring block */
#define ITERS   (DPB / 64)          /* 16 iterations of 64 docs */
#define QA_ELEMS (4 * 12 * 32 * 16) /* packed query bf16 elements = 24576 (48KB) */

// --------------------------------------------------------------------------
// Kernel 0: L2-normalize queries (f32), pack into WMMA bf16 A-operand layout.
// A-layout (16x32 MxK, wave32): lane = m + 16*hi ; element j -> k offset:
//   hi=0: j<8 -> j,      j>=8 -> j+8
//   hi=1: j<8 -> j+8,    j>=8 -> j+16
// --------------------------------------------------------------------------
__global__ __launch_bounds__(128) void nq_pack(const float* __restrict__ q,
                                               __bf16* __restrict__ qA) {
  __shared__ float red[128];
  const int b = blockIdx.x;
  const int t = threadIdx.x;
  const float* qr = q + (size_t)b * EMBED;
  float ss = 0.f;
  for (int k = t; k < EMBED; k += 128) { float v = qr[k]; ss += v * v; }
  red[t] = ss;
  __syncthreads();
  for (int off = 64; off > 0; off >>= 1) {
    if (t < off) red[t] += red[t + off];
    __syncthreads();
  }
  const float scale = 1.f / (sqrtf(red[0]) + 1e-8f);
  const int mtile = b >> 4, m = b & 15;
  for (int k = t; k < EMBED; k += 128) {
    const float v = qr[k] * scale;
    const int ks = k >> 5, ko = k & 31;
    const int hi = (ko >> 3) & 1;
    const int j  = (ko & 7) + ((ko >> 4) << 3);
    const int lane = m + (hi << 4);
    qA[((((mtile * 12 + ks) << 5) + lane) << 4) + j] = (__bf16)v;
  }
}

// --------------------------------------------------------------------------
// Kernel 1: fused doc-norm + bf16 WMMA scoring + block-local top-8.
// 128 threads = 4 waves; each wave owns a 16-doc tile per iteration.
// --------------------------------------------------------------------------
__global__ __launch_bounds__(128) void score_topk(const float* __restrict__ docs,
                                                  const __bf16* __restrict__ qA,
                                                  float* __restrict__ candS,
                                                  int*   __restrict__ candI) {
  __shared__ __align__(32) __bf16 sA[QA_ELEMS];  // 48 KB packed queries
  __shared__ float sScore[64 * 17];              // padded transpose buffer

  const int t    = threadIdx.x;
  const int lane = t & 31;
  const int wave = t >> 5;
  const int hi   = lane >> 4;       // which K-half this lane holds
  const int dcol = lane & 15;       // doc column within tile

  // Cooperative copy of packed queries global -> LDS (reused for 1024 docs).
  {
    const uint4* src = (const uint4*)qA;
    uint4* dst = (uint4*)sA;
    for (int i = t; i < (QA_ELEMS * 2) / 16; i += 128) dst[i] = src[i];
  }
  __syncthreads();

  float ts[KSEL];
  int   ti[KSEL];
#pragma unroll
  for (int j = 0; j < KSEL; ++j) { ts[j] = -1e30f; ti[j] = 0; }

  const int docBase = blockIdx.x * DPB;

  for (int it = 0; it < ITERS; ++it) {
    const int doc = docBase + it * 64 + wave * 16 + dcol;
    const float* dp = docs + (size_t)doc * EMBED + (hi << 3);

    float ss = 0.f;
    v8f acc0 = {}, acc1 = {}, acc2 = {}, acc3 = {};

#pragma unroll
    for (int ks = 0; ks < 12; ++ks) {
      const float4 f0 = *(const float4*)(dp + ks * 32 + 0);
      const float4 f1 = *(const float4*)(dp + ks * 32 + 4);
      const float4 f2 = *(const float4*)(dp + ks * 32 + 16);
      const float4 f3 = *(const float4*)(dp + ks * 32 + 20);

      ss += f0.x*f0.x + f0.y*f0.y + f0.z*f0.z + f0.w*f0.w;
      ss += f1.x*f1.x + f1.y*f1.y + f1.z*f1.z + f1.w*f1.w;
      ss += f2.x*f2.x + f2.y*f2.y + f2.z*f2.z + f2.w*f2.w;
      ss += f3.x*f3.x + f3.y*f3.y + f3.z*f3.z + f3.w*f3.w;

      v16bf bm;
      bm[0]  = (__bf16)f0.x; bm[1]  = (__bf16)f0.y; bm[2]  = (__bf16)f0.z; bm[3]  = (__bf16)f0.w;
      bm[4]  = (__bf16)f1.x; bm[5]  = (__bf16)f1.y; bm[6]  = (__bf16)f1.z; bm[7]  = (__bf16)f1.w;
      bm[8]  = (__bf16)f2.x; bm[9]  = (__bf16)f2.y; bm[10] = (__bf16)f2.z; bm[11] = (__bf16)f2.w;
      bm[12] = (__bf16)f3.x; bm[13] = (__bf16)f3.y; bm[14] = (__bf16)f3.z; bm[15] = (__bf16)f3.w;

      const v16bf a0 = *(const v16bf*)&sA[(((0 * 12 + ks) << 5) + lane) << 4];
      const v16bf a1 = *(const v16bf*)&sA[(((1 * 12 + ks) << 5) + lane) << 4];
      const v16bf a2 = *(const v16bf*)&sA[(((2 * 12 + ks) << 5) + lane) << 4];
      const v16bf a3 = *(const v16bf*)&sA[(((3 * 12 + ks) << 5) + lane) << 4];

      acc0 = __builtin_amdgcn_wmma_f32_16x16x32_bf16(false, a0, false, bm, (short)0, acc0, false, false);
      acc1 = __builtin_amdgcn_wmma_f32_16x16x32_bf16(false, a1, false, bm, (short)0, acc1, false, false);
      acc2 = __builtin_amdgcn_wmma_f32_16x16x32_bf16(false, a2, false, bm, (short)0, acc2, false, false);
      acc3 = __builtin_amdgcn_wmma_f32_16x16x32_bf16(false, a3, false, bm, (short)0, acc3, false, false);
    }

    // Each lane holds half the dims of its doc; partner lane is lane^16.
    const float ssF = ss + __shfl_xor(ss, 16);
    const float dscale = 1.f / (sqrtf(ssF) + 1e-8f);
    const int qhi = hi << 3;  // C/D layout: VGPR r -> M = r + 8*hi

    // Sub-rounds: one wave at a time transposes its 16-doc scores through LDS,
    // then 64 query-owner threads update register-resident sorted top-8.
    for (int w = 0; w < 4; ++w) {
      if (wave == w) {
#pragma unroll
        for (int r = 0; r < 8; ++r) {
          sScore[( 0 + r + qhi) * 17 + dcol] = acc0[r] * dscale;
          sScore[(16 + r + qhi) * 17 + dcol] = acc1[r] * dscale;
          sScore[(32 + r + qhi) * 17 + dcol] = acc2[r] * dscale;
          sScore[(48 + r + qhi) * 17 + dcol] = acc3[r] * dscale;
        }
      }
      __syncthreads();
      if (t < 64) {
        const int idxBase = docBase + it * 64 + w * 16;
        for (int d = 0; d < 16; ++d) {
          const float s = sScore[t * 17 + d];
          if (s > ts[KSEL - 1]) {          // rare path after warmup
            float cs = s; int ci = idxBase + d;
#pragma unroll
            for (int j = 0; j < KSEL; ++j) {   // branch-free sorted bubble
              const bool g = cs > ts[j];
              const float tf = ts[j]; const int tx = ti[j];
              ts[j] = g ? cs : tf;  ti[j] = g ? ci : tx;
              cs    = g ? tf : cs;  ci    = g ? tx : ci;
            }
          }
        }
      }
      __syncthreads();
    }
  }

  if (t < 64) {
    const size_t base = ((size_t)blockIdx.x * 64 + t) * KSEL;
#pragma unroll
    for (int j = 0; j < KSEL; ++j) { candS[base + j] = ts[j]; candI[base + j] = ti[j]; }
  }
}

// --------------------------------------------------------------------------
// Kernel 2: per-query merge of 256 sorted 8-lists -> global top-8,
// softmax weights, renormalized gather + fusion. One block per query.
// --------------------------------------------------------------------------
__global__ __launch_bounds__(256) void reduce_fuse(const float* __restrict__ docs,
                                                   const float* __restrict__ candS,
                                                   const int*   __restrict__ candI,
                                                   float* __restrict__ outFused,
                                                   float* __restrict__ outScore,
                                                   float* __restrict__ outIdx) {
  __shared__ float sVal[256];
  __shared__ int   sOwn[256];
  __shared__ int   sDoc[256];
  __shared__ float selS[KSEL];
  __shared__ int   selI[KSEL];
  __shared__ float eW[KSEL];
  __shared__ float sScale[KSEL];
  __shared__ int   sWin;

  const int b = blockIdx.x;
  const int t = threadIdx.x;

  float h[KSEL]; int hx[KSEL];
  const size_t cbase = ((size_t)t * 64 + b) * KSEL;
#pragma unroll
  for (int j = 0; j < KSEL; ++j) { h[j] = candS[cbase + j]; hx[j] = candI[cbase + j]; }

  for (int iter = 0; iter < KSEL; ++iter) {
    sVal[t] = h[0]; sOwn[t] = t; sDoc[t] = hx[0];
    __syncthreads();
    for (int off = 128; off > 0; off >>= 1) {
      if (t < off && sVal[t + off] > sVal[t]) {
        sVal[t] = sVal[t + off]; sOwn[t] = sOwn[t + off]; sDoc[t] = sDoc[t + off];
      }
      __syncthreads();
    }
    if (t == 0) { selS[iter] = sVal[0]; selI[iter] = sDoc[0]; sWin = sOwn[0]; }
    __syncthreads();
    if (t == sWin) {   // winner pops its sorted head
#pragma unroll
      for (int j = 0; j < KSEL - 1; ++j) { h[j] = h[j + 1]; hx[j] = hx[j + 1]; }
      h[KSEL - 1] = -1e30f;
    }
    __syncthreads();
  }

  // Softmax over the (descending) top-8 scores.
  if (t < KSEL) eW[t] = __expf(selS[t] - selS[0]);
  __syncthreads();
  float wsum = 0.f;
#pragma unroll
  for (int j = 0; j < KSEL; ++j) wsum += eW[j];

  // 8 waves compute the 8 selected doc norms (f32).
  {
    const int w = t >> 5, ln = t & 31;
    const float* dr = docs + (size_t)selI[w] * EMBED;
    float ss = 0.f;
#pragma unroll
    for (int s = 0; s < 12; ++s) { const float v = dr[ln + 32 * s]; ss += v * v; }
#pragma unroll
    for (int off = 16; off > 0; off >>= 1) ss += __shfl_xor(ss, off);
    if (ln == 0) sScale[w] = 1.f / (sqrtf(ss) + 1e-8f);
  }
  __syncthreads();

  for (int dim = t; dim < EMBED; dim += 256) {
    float acc = 0.f;
#pragma unroll
    for (int j = 0; j < KSEL; ++j)
      acc += eW[j] * sScale[j] * docs[(size_t)selI[j] * EMBED + dim];
    outFused[(size_t)b * EMBED + dim] = acc / wsum;
  }
  if (t < KSEL) {
    outScore[b * KSEL + t] = selS[t];
    outIdx  [b * KSEL + t] = (float)selI[t];
  }
}

// --------------------------------------------------------------------------
extern "C" void kernel_launch(void* const* d_in, const int* in_sizes, int n_in,
                              void* d_out, int out_size, void* d_ws, size_t ws_size,
                              hipStream_t stream) {
  (void)in_sizes; (void)n_in; (void)out_size; (void)ws_size;
  const float* q    = (const float*)d_in[0];
  const float* docs = (const float*)d_in[1];
  // d_in[2] = top_k scalar (fixed at 8, compiled in)

  __bf16* qA   = (__bf16*)d_ws;                                   // 48 KB
  float*  candS = (float*)((char*)d_ws + QA_ELEMS * 2);           // 512 KB
  int*    candI = (int*)((char*)d_ws + QA_ELEMS * 2
                         + (size_t)SBLOCKS * 64 * KSEL * 4);      // 512 KB

  float* outF = (float*)d_out;                 // [64][384]
  float* outS = outF + BATCH * EMBED;          // [64][8]
  float* outI = outS + BATCH * KSEL;           // [64][8] (indices as float)

  hipLaunchKernelGGL(nq_pack,     dim3(BATCH),   dim3(128), 0, stream, q, qA);
  hipLaunchKernelGGL(score_topk,  dim3(SBLOCKS), dim3(128), 0, stream, docs, qA, candS, candI);
  hipLaunchKernelGGL(reduce_fuse, dim3(BATCH),   dim3(256), 0, stream, docs, candS, candI,
                     outF, outS, outI);
}
